// RPE_81879256531612
// MI455X (gfx1250) — matrix operand order repaired and checked by reading, not compiled, not run on hardware
//
#include <hip/hip_runtime.h>
#include <hip/hip_bf16.h>
#include <math.h>
#include <stdint.h>

typedef __attribute__((ext_vector_type(16))) _Float16 v16h;
typedef __attribute__((ext_vector_type(8)))  _Float16 v8h;
typedef __attribute__((ext_vector_type(8)))  float    v8f;

#define Bb   2
#define Tt   128
#define Dd   256
#define Hh   8
#define Ff   64
#define Cc   512
#define TEMBK 512

// ---- CDNA5 async-copy helpers (ASYNCcnt path, inline asm: the clang builtin
//      wants LangAS::cuda_device pointers which HIP source cannot spell) ----
__device__ __forceinline__ void async_load_lds_b128(void* lds_ptr, const void* gaddr) {
  // Generic LDS-aperture address keeps the wave-relative LDS byte offset in
  // addr[31:0] (ISA flat-aperture rules), which is exactly what VDST needs.
  uint32_t lds_off = (uint32_t)(uintptr_t)lds_ptr;
  asm volatile("global_load_async_to_lds_b128 %0, %1, off"
               :: "v"(lds_off), "v"(gaddr)
               : "memory");
}

__device__ __forceinline__ void wait_async0() {
#if __has_builtin(__builtin_amdgcn_s_wait_asynccnt)
  __builtin_amdgcn_s_wait_asynccnt(0);
#else
  asm volatile("s_wait_asynccnt 0" ::: "memory");
#endif
}

// ---- WMMA fragment loaders (wave32; contiguous K order, same for A and B) ----

// A matrix 16x32 f16 from row-major f16 [rows][stride]:
// lane l: M = l&15; 16 contiguous K starting at 16*(l>>4)
__device__ __forceinline__ v16h load_a_f16(const _Float16* src, int stride,
                                           int row0, int col0, int lane) {
  int m  = lane & 15;
  int kh = (lane >> 4) << 4;            // 0 or 16
  const _Float16* p = src + (size_t)(row0 + m) * stride + col0 + kh;
  union { v16h v; v8h h[2]; } u;
  u.h[0] = *(const v8h*)(p);            // K = col0+kh .. +7
  u.h[1] = *(const v8h*)(p + 8);        // K = col0+kh+8 .. +15
  return u.v;
}

// B matrix 32x16 f16 where the SOURCE is stored [N][K] row-major (i.e. B^T):
// lane l: N = l&15; 16 contiguous K starting at 16*(l>>4)
__device__ __forceinline__ v16h load_b_f16(const _Float16* srcT, int stride,
                                           int n0, int k0, int lane) {
  const _Float16* p = srcT + (size_t)(n0 + (lane & 15)) * stride
                           + k0 + ((lane >> 4) << 4);
  return *(const v16h*)(p);
}

// ---- Kernel A: temb_emb[b,t,c] = temb[b,t,:] . w_time[c,:] + b_time[c] ----
__global__ void __launch_bounds__(256)
temb_gemm_kernel(const float* __restrict__ temb, const float* __restrict__ w_time,
                 const float* __restrict__ b_time, float* __restrict__ temb_emb) {
  int t = blockIdx.x, b = blockIdx.y, tid = threadIdx.x;
  __shared__ float trow[TEMBK];
  for (int i = tid; i < TEMBK; i += 256) trow[i] = temb[((size_t)b * Tt + t) * TEMBK + i];
  __syncthreads();
  for (int c = tid; c < Cc; c += 256) {
    const float* wr = w_time + (size_t)c * TEMBK;
    float acc = 0.f;
#pragma unroll 4
    for (int k = 0; k < TEMBK; ++k) acc = fmaf(trow[k], wr[k], acc);
    temb_emb[((size_t)b * Tt + t) * Cc + c] = acc + b_time[c];
  }
}

// ---- Kernel W: f32 -> f16 convert (for w_out) ----
__global__ void cvt_f16_kernel(const float* __restrict__ src,
                               _Float16* __restrict__ dst, int n) {
  int i = blockIdx.x * blockDim.x + threadIdx.x;
  if (i < n) dst[i] = (_Float16)src[i];
}

// ---- Kernel B: R[b,t,s,o] = silu(emb) @ w_out^T + b_out  (WMMA f16) ----
// Grid: (4 s-blocks of 32, T, B); block 256 = 8 wave32.
__global__ void __launch_bounds__(256)
rpe_gemm_kernel(const float* __restrict__ pd,
                const float* __restrict__ w_dist, const float* __restrict__ b_dist,
                const float* __restrict__ b_out,
                const float* __restrict__ temb_emb,
                const _Float16* __restrict__ w_out_h,
                _Float16* __restrict__ R) {
  int sb = blockIdx.x, t = blockIdx.y, b = blockIdx.z, tid = threadIdx.x;

  __shared__ alignas(32) _Float16 act[32 * Cc];   // 32 KB silu activations
  __shared__ float df[32][3];
  __shared__ float baseC[Cc];
  __shared__ float wd[Cc * 3];

  if (tid < 32) {
    float p = pd[((size_t)b * Tt + t) * Tt + sb * 32 + tid];
    df[tid][0] = log1pf(fmaxf(p, 0.f));
    df[tid][1] = log1pf(fmaxf(-p, 0.f));
    df[tid][2] = (p == 0.f) ? 1.f : 0.f;
  }
  for (int c = tid; c < Cc; c += 256)
    baseC[c] = temb_emb[((size_t)b * Tt + t) * Cc + c] + b_dist[c];
  for (int i = tid; i < Cc * 3; i += 256) wd[i] = w_dist[i];
  __syncthreads();

  for (int i = tid; i < 32 * Cc; i += 256) {
    int s = i >> 9, c = i & (Cc - 1);
    float e = baseC[c] + df[s][0] * wd[c * 3] + df[s][1] * wd[c * 3 + 1]
                       + df[s][2] * wd[c * 3 + 2];
    float a = e / (1.f + __expf(-e));     // silu
    act[s * Cc + c] = (_Float16)a;
  }
  __syncthreads();

  int wave = tid >> 5, lane = tid & 31;
  int mt   = wave & 1;                    // which 16-row m tile of the 32
  int ngrp = (wave >> 1) * 8;             // 8 n-tiles per wave

  for (int np = 0; np < 8; np += 2) {     // two n-tiles at a time (A reuse)
    int n0 = (ngrp + np) * 16, n1 = n0 + 16;
    float bias0 = b_out[n0 + (lane & 15)];
    float bias1 = b_out[n1 + (lane & 15)];
    v8f acc0, acc1;
    for (int j = 0; j < 8; ++j) { acc0[j] = bias0; acc1[j] = bias1; }
    for (int k0 = 0; k0 < Cc; k0 += 32) {
      v16h a  = load_a_f16(act, Cc, mt * 16, k0, lane);
      v16h f0 = load_b_f16(w_out_h, Cc, n0, k0, lane);
      v16h f1 = load_b_f16(w_out_h, Cc, n1, k0, lane);
      acc0 = __builtin_amdgcn_wmma_f32_16x16x32_f16(false, a, false, f0, (short)0, acc0, false, false);
      acc1 = __builtin_amdgcn_wmma_f32_16x16x32_f16(false, a, false, f1, (short)0, acc1, false, false);
    }
    int srow = sb * 32 + mt * 16 + ((lane >> 4) << 3);   // M = j + 8*(lane>=16)
    size_t base = (((size_t)b * Tt + t) * Tt + srow) * Cc;
    int o0 = n0 + (lane & 15), o1 = n1 + (lane & 15);
    for (int j = 0; j < 8; ++j) {
      R[base + (size_t)j * Cc + o0] = (_Float16)acc0[j];
      R[base + (size_t)j * Cc + o1] = (_Float16)acc1[j];
    }
  }
}

// ---- Kernel C: out[b,d,h,t,s] = sum_f qk[b,d,h,t,f] * R[b,t,s,h*64+f] ----
// Grid: (T, H, B); per-block GEMM M=256(d) x N=128(s) x K=64(f) from LDS.
__global__ void __launch_bounds__(256)
qkr_gemm_kernel(const float* __restrict__ qk, const _Float16* __restrict__ R,
                float* __restrict__ out) {
  int t = blockIdx.x, h = blockIdx.y, b = blockIdx.z, tid = threadIdx.x;
  __shared__ alignas(32) _Float16 qkL[Dd * Ff];   // 32 KB [d][f]
  __shared__ alignas(32) _Float16 rL[Tt * Ff];    // 16 KB [s][f]

  // R tile: already f16 in memory -> CDNA5 async Memory->LDS copy (ASYNCcnt),
  // overlapping with the qk f32->f16 VALU staging loop below.
#if defined(__gfx1250__)
  for (int i = tid; i < (Tt * Ff) / 8; i += 256) {     // 16B (8 halves) per chunk
    int s = i >> 3, f = (i & 7) * 8;
    const _Float16* gp = R + (((size_t)b * Tt + t) * Tt + s) * Cc + h * Ff + f;
    async_load_lds_b128(&rL[i * 8], gp);
  }
#else
  for (int i = tid; i < Tt * Ff; i += 256) {
    int s = i >> 6, f = i & 63;
    rL[i] = R[(((size_t)b * Tt + t) * Tt + s) * Cc + h * Ff + f];
  }
#endif

  for (int i = tid; i < Dd * Ff; i += 256) {
    int d = i >> 6, f = i & 63;
    qkL[i] = (_Float16)qk[((((size_t)b * Dd + d) * Hh + h) * Tt + t) * Ff + f];
  }
  wait_async0();            // async loads are NOT covered by the barrier's LDS wait
  __syncthreads();

  int wave = tid >> 5, lane = tid & 31;
  for (int mi = 0; mi < 2; ++mi) {
    int mt = wave * 2 + mi;
    v16h a0 = load_a_f16(qkL, Ff, mt * 16, 0, lane);
    v16h a1 = load_a_f16(qkL, Ff, mt * 16, 32, lane);
    int drow = mt * 16 + ((lane >> 4) << 3);
    for (int nt = 0; nt < 8; ++nt) {
      v16h b0 = load_b_f16(rL, Ff, nt * 16, 0, lane);
      v16h b1 = load_b_f16(rL, Ff, nt * 16, 32, lane);
      v8f acc = {};
      acc = __builtin_amdgcn_wmma_f32_16x16x32_f16(false, a0, false, b0, (short)0, acc, false, false);
      acc = __builtin_amdgcn_wmma_f32_16x16x32_f16(false, a1, false, b1, (short)0, acc, false, false);
      int scol = nt * 16 + (lane & 15);
      size_t obase = ((((size_t)b * Dd + drow) * Hh + h) * Tt + t) * Tt + scol;
      for (int j = 0; j < 8; ++j)
        out[obase + (size_t)j * (Hh * Tt * Tt)] = acc[j];
    }
  }
}

extern "C" void kernel_launch(void* const* d_in, const int* in_sizes, int n_in,
                              void* d_out, int out_size, void* d_ws, size_t ws_size,
                              hipStream_t stream) {
  (void)in_sizes; (void)n_in; (void)out_size; (void)ws_size;
  const float* qk     = (const float*)d_in[0];
  const float* pd     = (const float*)d_in[1];
  const float* temb   = (const float*)d_in[2];
  const float* w_dist = (const float*)d_in[3];
  const float* b_dist = (const float*)d_in[4];
  const float* w_time = (const float*)d_in[5];
  const float* b_time = (const float*)d_in[6];
  const float* w_out  = (const float*)d_in[7];
  const float* b_out  = (const float*)d_in[8];
  float* out = (float*)d_out;

  char* ws = (char*)d_ws;
  float*    temb_emb = (float*)ws;                                        // 512 KB
  _Float16* w_out_h  = (_Float16*)(ws + (size_t)Bb * Tt * Cc * 4);        // 512 KB
  _Float16* R        = (_Float16*)(ws + (size_t)Bb * Tt * Cc * 4
                                      + (size_t)Cc * Cc * 2);             // 32 MB

  temb_gemm_kernel<<<dim3(Tt, Bb), 256, 0, stream>>>(temb, w_time, b_time, temb_emb);
  cvt_f16_kernel<<<(Cc * Cc + 255) / 256, 256, 0, stream>>>(w_out, w_out_h, Cc * Cc);
  rpe_gemm_kernel<<<dim3(4, Tt, Bb), 256, 0, stream>>>(pd, w_dist, b_dist, b_out,
                                                       temb_emb, w_out_h, R);
  qkr_gemm_kernel<<<dim3(Tt, Hh, Bb), 256, 0, stream>>>(qk, R, out);
}
